// Net_63196148793448
// MI455X (gfx1250) — matrix operand order repaired and checked by reading, not compiled, not run on hardware
//
#include <hip/hip_runtime.h>
#include <cstdint>
#include <cstddef>

// ---------------------------------------------------------------- types
typedef __attribute__((ext_vector_type(16))) _Float16 v16h;
typedef __attribute__((ext_vector_type(8)))  float    v8f;
typedef __attribute__((ext_vector_type(2)))  _Float16 h2;

union Frag16 { uint4 q[2]; v16h v; };   // 32 bytes: 8 VGPRs of packed halves

// ---------------------------------------------------------------- problem dims
#define NN   8192     // nodes
#define NE   32768    // edges
#define NB   256      // graphs
#define DD   64       // hidden dim
#define HH   128      // edge-MLP hidden
#define FIN  11
#define EIN  5
#define DF   4096     // D*D

__device__ __forceinline__ float sigm(float x) { return 1.0f / (1.0f + expf(-x)); }

// ---------------------------------------------------------------- utility
__global__ void zero_f32(float* __restrict__ p, int n) {
  int i = blockIdx.x * blockDim.x + threadIdx.x;
  if (i < n) p[i] = 0.0f;
}

// ---------------------------------------------------------------- edge hidden: A = relu(edge_attr @ nn1_w^T + nn1_b) -> f16 [E,128]
__global__ __launch_bounds__(256)
void edge_hidden(const float* __restrict__ ea, const float* __restrict__ w,
                 const float* __restrict__ b, _Float16* __restrict__ Ah) {
  int t = blockIdx.x * 256 + threadIdx.x;     // E*128 threads
  int e = t >> 7, h = t & 127;
  const float* a  = ea + (size_t)e * EIN;
  const float* wr = w + (size_t)h * EIN;
  float v = b[h];
#pragma unroll
  for (int i = 0; i < EIN; ++i) v += a[i] * wr[i];
  Ah[(size_t)e * HH + h] = (_Float16)fmaxf(v, 0.0f);
}

// ---------------------------------------------------------------- Bm[h][df] = (f16) nn2_w[df][h]   (transpose to [K,N] row-major)
__global__ __launch_bounds__(256)
void prep_b(const float* __restrict__ nn2w, _Float16* __restrict__ Bm) {
  int t = blockIdx.x * 256 + threadIdx.x;     // 128*4096 threads
  int h = t >> 12, df = t & 4095;
  Bm[(size_t)h * DF + df] = (_Float16)nn2w[(size_t)df * HH + h];
}

// ---------------------------------------------------------------- WMMA GEMM: We[E,4096] = Ah[E,128] @ Bm[128,4096] + nn2_b, f16 out
#define BM 128
#define BN 128
#define KTOT 128
#define LDSA (KTOT + 8)   // halves per row, pad 8 (row stride 272B, 16B aligned)

__global__ __launch_bounds__(256)
void we_gemm(const _Float16* __restrict__ A, const _Float16* __restrict__ B,
             const float* __restrict__ bias, _Float16* __restrict__ C) {
  __shared__ __align__(16) _Float16 As[BM * LDSA];   // 34816 B
  __shared__ __align__(16) _Float16 Bs[BN * LDSA];   // 34816 B  (stored [n][k])

  const int tid = threadIdx.x;
  const int m0 = blockIdx.y * BM;
  const int n0 = blockIdx.x * BN;

  // ---- stage A tile: 128 rows x 128 halves, 256B per row, 2 threads/row
  {
    int row = tid >> 1;
    int c0  = (tid & 1) * 64;                       // 64 halves = 128B
    const uint4* gs = (const uint4*)(A + (size_t)(m0 + row) * KTOT + c0);
    uint4* ls = (uint4*)(As + row * LDSA + c0);
#pragma unroll
    for (int j = 0; j < 8; ++j) ls[j] = gs[j];
  }
  // ---- stage B tile transposed: Bs[n][k] = B[k][n0+n]
  {
    int k  = tid >> 1;
    int c0 = (tid & 1) * 64;
    const _Float16* gs = B + (size_t)k * DF + n0 + c0;
#pragma unroll 8
    for (int j = 0; j < 64; ++j) Bs[(c0 + j) * LDSA + k] = gs[j];
  }
  __syncthreads();

  const int lane = tid & 31;
  const int wave = tid >> 5;
  const int wr = wave & 3;      // rows  wr*32 .. +31 (two 16-row tiles)
  const int wc = wave >> 2;     // cols  wc*64 .. +63 (four 16-col tiles)
  const int l16 = lane & 15;
  const int hi  = lane >> 4;    // half-wave select

  v8f zero8 = {0.f, 0.f, 0.f, 0.f, 0.f, 0.f, 0.f, 0.f};
  v8f acc[2][4];
#pragma unroll
  for (int s = 0; s < 2; ++s)
#pragma unroll
    for (int t = 0; t < 4; ++t) acc[s][t] = zero8;

#pragma unroll
  for (int ks = 0; ks < 4; ++ks) {
    const int kofs = ks * 32;                       // halves
    // A fragments: ISA 16-bit A 16x32 layout.
    // lane<16: VGPR r<4 -> K=2r,2r+1 ; r>=4 -> K=16+2(r-4)..   lane>=16: +8
    Frag16 af[2];
#pragma unroll
    for (int s = 0; s < 2; ++s) {
      const unsigned int* ar =
          (const unsigned int*)(As + (wr * 32 + s * 16 + l16) * LDSA) + (kofs >> 1);
      af[s].q[0] = *(const uint4*)(ar + 4 * hi);
      af[s].q[1] = *(const uint4*)(ar + 8 + 4 * hi);
    }
    // B fragments: same K order as A (lane<16: K=0..7,16..23 ; lane>=16: K=8..15,24..31), col N=l16
    Frag16 bf[4];
#pragma unroll
    for (int t = 0; t < 4; ++t) {
      const unsigned int* br =
          (const unsigned int*)(Bs + (wc * 64 + t * 16 + l16) * LDSA) + (kofs >> 1);
      bf[t].q[0] = *(const uint4*)(br + 4 * hi);
      bf[t].q[1] = *(const uint4*)(br + 8 + 4 * hi);
    }
#pragma unroll
    for (int s = 0; s < 2; ++s)
#pragma unroll
      for (int t = 0; t < 4; ++t)
        acc[s][t] = __builtin_amdgcn_wmma_f32_16x16x32_f16(
            false, af[s].v, false, bf[t].v, (short)0, acc[s][t], false, false);
  }

  // ---- store: C/D layout VGPR v -> M = v + 8*hi, N = l16
#pragma unroll
  for (int s = 0; s < 2; ++s)
#pragma unroll
    for (int t = 0; t < 4; ++t) {
      const int gn = n0 + wc * 64 + t * 16 + l16;
      const float bv = bias[gn];
      const int gm0 = m0 + wr * 32 + s * 16 + 8 * hi;
#pragma unroll
      for (int v = 0; v < 8; ++v)
        C[(size_t)(gm0 + v) * DF + gn] = (_Float16)(acc[s][t][v] + bv);
    }
}

// ---------------------------------------------------------------- degree + inverse
__global__ void deg_accum(const int* __restrict__ ei, int* __restrict__ deg) {
  int e = blockIdx.x * blockDim.x + threadIdx.x;
  if (e < NE) atomicAdd(&deg[ei[NE + e]], 1);
}
__global__ void make_invdeg(const int* __restrict__ deg, float* __restrict__ invd) {
  int n = blockIdx.x * blockDim.x + threadIdx.x;
  if (n < NN) { int d = deg[n]; invd[n] = d > 0 ? 1.0f / (float)d : 0.0f; }
}

// ---------------------------------------------------------------- lin0: h = relu(x @ lin0_w^T + b)
__global__ __launch_bounds__(256)
void lin0_kernel(const float* __restrict__ x, const float* __restrict__ w,
                 const float* __restrict__ b, float* __restrict__ hbuf) {
  int t = blockIdx.x * 256 + threadIdx.x;     // N*64 threads
  int n = t >> 6, f = t & 63;
  const float* xr = x + (size_t)n * FIN;
  const float* wr = w + (size_t)f * FIN;
  float v = b[f];
#pragma unroll
  for (int i = 0; i < FIN; ++i) v += xr[i] * wr[i];
  hbuf[t] = fmaxf(v, 0.0f);
}

// ---------------------------------------------------------------- msg + scatter: one wave per edge, lane owns features (2l,2l+1)
__global__ __launch_bounds__(256)
void msg_scatter(const _Float16* __restrict__ We, const float* __restrict__ hbuf,
                 const int* __restrict__ ei, float* __restrict__ agg) {
  const int wave = threadIdx.x >> 5;
  const int lane = threadIdx.x & 31;
  const int e = blockIdx.x * 8 + wave;
  const int src = ei[e];
  const int dst = ei[NE + e];
  const h2* wp = (const h2*)(We + (size_t)e * DF);   // [d][32 pairs], 128B per d, coalesced
  const float* xs = hbuf + (size_t)src * DD;
  float a0 = 0.0f, a1 = 0.0f;
#pragma unroll 8
  for (int d = 0; d < DD; ++d) {
    h2 w = wp[d * 32 + lane];
    float xd = xs[d];
    a0 += xd * (float)w[0];
    a1 += xd * (float)w[1];
  }
  atomicAdd(&agg[(size_t)dst * DD + 2 * lane],     a0);
  atomicAdd(&agg[(size_t)dst * DD + 2 * lane + 1], a1);
}

// ---------------------------------------------------------------- NNConv combine + GRU step (one 64-thread block per node)
__global__ __launch_bounds__(64)
void node_update(float* __restrict__ hbuf, const float* __restrict__ agg,
                 const float* __restrict__ invd, const float* __restrict__ rootw,
                 const float* __restrict__ convb, const float* __restrict__ wih,
                 const float* __restrict__ whh, const float* __restrict__ bih,
                 const float* __restrict__ bhh) {
  const int n = blockIdx.x, f = threadIdx.x;
  __shared__ float hs[DD], ms[DD];
  hs[f] = hbuf[(size_t)n * DD + f];
  __syncthreads();

  float m = agg[(size_t)n * DD + f] * invd[n] + convb[f];
  const float* rw = rootw + (size_t)f * DD;
#pragma unroll 8
  for (int d = 0; d < DD; ++d) m += hs[d] * rw[d];
  m = fmaxf(m, 0.0f);
  ms[f] = m;
  __syncthreads();

  float xr = bih[f], xz = bih[64 + f], xn = bih[128 + f];
  float hr = bhh[f], hz = bhh[64 + f], hn = bhh[128 + f];
  const float* wir = wih + (size_t)f * DD;
  const float* wiz = wih + (size_t)(64 + f) * DD;
  const float* win = wih + (size_t)(128 + f) * DD;
  const float* whr = whh + (size_t)f * DD;
  const float* whz = whh + (size_t)(64 + f) * DD;
  const float* whn = whh + (size_t)(128 + f) * DD;
#pragma unroll 4
  for (int d = 0; d < DD; ++d) {
    float md = ms[d], hd = hs[d];
    xr += md * wir[d];  xz += md * wiz[d];  xn += md * win[d];
    hr += hd * whr[d];  hz += hd * whz[d];  hn += hd * whn[d];
  }
  float r = sigm(xr + hr);
  float z = sigm(xz + hz);
  float nn = tanhf(xn + r * hn);
  hbuf[(size_t)n * DD + f] = (1.0f - z) * nn + z * hs[f];
}

// ---------------------------------------------------------------- per-graph node ranges (batch is sorted)
__global__ void graph_bounds(const int* __restrict__ batch, int* __restrict__ gstart) {
  int n = blockIdx.x * blockDim.x + threadIdx.x;
  if (n > NN) return;
  int bn = (n < NN) ? batch[n] : NB;
  int bp = (n == 0) ? -1 : batch[n - 1];
  for (int g = bp + 1; g <= bn; ++g) gstart[g] = n;
}

// ---------------------------------------------------------------- one Set2Set step (one 64-thread block per graph)
__global__ __launch_bounds__(64)
void s2s_step(const float* __restrict__ hbuf, const int* __restrict__ gstart,
              const float* __restrict__ wih, const float* __restrict__ whh,
              const float* __restrict__ bih, const float* __restrict__ bhh,
              float* __restrict__ qh, float* __restrict__ qc, float* __restrict__ qstar) {
  const int b = blockIdx.x, f = threadIdx.x;
  __shared__ float qs[2 * DD], qhs[DD], red[DD], racc[DD];

  qhs[f]      = qh[(size_t)b * DD + f];
  qs[f]       = qstar[(size_t)b * 2 * DD + f];
  qs[DD + f]  = qstar[(size_t)b * 2 * DD + DD + f];
  __syncthreads();

  // LSTM gates: rows [i,f,g,o] of s2s_w_ih [256,128] and s2s_w_hh [256,64]
  float g[4];
#pragma unroll
  for (int gi = 0; gi < 4; ++gi) {
    int row = gi * DD + f;
    float a = bih[row] + bhh[row];
    const float* wi = wih + (size_t)row * (2 * DD);
#pragma unroll 4
    for (int j = 0; j < 2 * DD; ++j) a += qs[j] * wi[j];
    const float* wh = whh + (size_t)row * DD;
#pragma unroll 4
    for (int j = 0; j < DD; ++j) a += qhs[j] * wh[j];
    g[gi] = a;
  }
  float qcv = sigm(g[1]) * qc[(size_t)b * DD + f] + sigm(g[0]) * tanhf(g[2]);
  float qhv = sigm(g[3]) * tanhf(qcv);
  qc[(size_t)b * DD + f] = qcv;
  qh[(size_t)b * DD + f] = qhv;
  __syncthreads();               // everyone done reading old qhs
  qhs[f]  = qhv;                 // q vector for attention
  racc[f] = 0.0f;
  __syncthreads();

  const int n0 = gstart[b], n1 = gstart[b + 1];

  // pass 1: segment max of e_n = <h_n, q>
  float lm = -3.4e38f;
  for (int n = n0 + f; n < n1; n += 64) {
    const float* hv = hbuf + (size_t)n * DD;
    float ea = 0.0f;
#pragma unroll 8
    for (int j = 0; j < DD; ++j) ea += hv[j] * qhs[j];
    lm = fmaxf(lm, ea);
  }
  red[f] = lm;
  __syncthreads();
  for (int s = 32; s > 0; s >>= 1) {
    if (f < s) red[f] = fmaxf(red[f], red[f + s]);
    __syncthreads();
  }
  float emax = red[0];
  if (!(emax > -3.0e38f)) emax = 0.0f;   // mimic where(isfinite(emax), emax, 0)
  __syncthreads();

  // pass 2: softmax weights + weighted sum of node features
  float wsum = 0.0f;
  for (int n = n0 + f; n < n1; n += 64) {
    const float* hv = hbuf + (size_t)n * DD;
    float ea = 0.0f;
#pragma unroll 8
    for (int j = 0; j < DD; ++j) ea += hv[j] * qhs[j];
    float wv = expf(ea - emax);
    wsum += wv;
#pragma unroll 4
    for (int j = 0; j < DD; ++j) atomicAdd(&racc[j], wv * hv[j]);   // ds_add_f32
  }
  red[f] = wsum;
  __syncthreads();
  for (int s = 32; s > 0; s >>= 1) {
    if (f < s) red[f] += red[f + s];
    __syncthreads();
  }
  float rn = 1.0f / fmaxf(red[0], 1e-12f);
  qstar[(size_t)b * 2 * DD + f]      = qhs[f];
  qstar[(size_t)b * 2 * DD + DD + f] = racc[f] * rn;
}

// ---------------------------------------------------------------- head: y = relu(q* @ l1^T + b1) @ l2^T + b2
__global__ __launch_bounds__(64)
void final_mlp(const float* __restrict__ qstar, const float* __restrict__ l1w,
               const float* __restrict__ l1b, const float* __restrict__ l2w,
               const float* __restrict__ l2b, float* __restrict__ y) {
  const int b = blockIdx.x, f = threadIdx.x;
  __shared__ float red[DD];
  const float* q = qstar + (size_t)b * 2 * DD;
  const float* w = l1w + (size_t)f * 2 * DD;
  float a = l1b[f];
#pragma unroll 4
  for (int j = 0; j < 2 * DD; ++j) a += q[j] * w[j];
  red[f] = fmaxf(a, 0.0f) * l2w[f];
  __syncthreads();
  for (int s = 32; s > 0; s >>= 1) {
    if (f < s) red[f] += red[f + s];
    __syncthreads();
  }
  if (f == 0) y[b] = red[0] + l2b[0];
}

// ================================================================ launcher
extern "C" void kernel_launch(void* const* d_in, const int* in_sizes, int n_in,
                              void* d_out, int out_size, void* d_ws, size_t ws_size,
                              hipStream_t stream) {
  const float* x      = (const float*)d_in[0];
  const float* ea     = (const float*)d_in[1];
  const float* lin0_w = (const float*)d_in[2];
  const float* lin0_b = (const float*)d_in[3];
  const float* nn1_w  = (const float*)d_in[4];
  const float* nn1_b  = (const float*)d_in[5];
  const float* nn2_w  = (const float*)d_in[6];
  const float* nn2_b  = (const float*)d_in[7];
  const float* root_w = (const float*)d_in[8];
  const float* conv_b = (const float*)d_in[9];
  const float* gw_ih  = (const float*)d_in[10];
  const float* gw_hh  = (const float*)d_in[11];
  const float* gb_ih  = (const float*)d_in[12];
  const float* gb_hh  = (const float*)d_in[13];
  const float* sw_ih  = (const float*)d_in[14];
  const float* sw_hh  = (const float*)d_in[15];
  const float* sb_ih  = (const float*)d_in[16];
  const float* sb_hh  = (const float*)d_in[17];
  const float* l1w    = (const float*)d_in[18];
  const float* l1b    = (const float*)d_in[19];
  const float* l2w    = (const float*)d_in[20];
  const float* l2b    = (const float*)d_in[21];
  const int*   ei     = (const int*)d_in[22];
  const int*   batch  = (const int*)d_in[23];
  float* y = (float*)d_out;

  char* ws = (char*)d_ws;
  size_t off = 0;
  auto alloc = [&](size_t bytes) -> void* {
    void* p = ws + off;
    off = (off + bytes + 255) & ~(size_t)255;
    return p;
  };

  _Float16* We  = (_Float16*)alloc((size_t)NE * DF * 2);     // 256 MB
  _Float16* Ah  = (_Float16*)alloc((size_t)NE * HH * 2);     // 8 MB
  _Float16* Bm  = (_Float16*)alloc((size_t)HH * DF * 2);     // 1 MB
  float* hbuf   = (float*)alloc((size_t)NN * DD * 4);        // 2 MB
  float* agg    = (float*)alloc((size_t)NN * DD * 4);        // 2 MB
  int*   deg    = (int*)alloc((size_t)NN * 4);
  float* invd   = (float*)alloc((size_t)NN * 4);
  int*   gstart = (int*)alloc((size_t)(NB + 1) * 4);
  float* qh     = (float*)alloc((size_t)NB * DD * 4);
  float* qc     = (float*)alloc((size_t)NB * DD * 4);
  float* qstar  = (float*)alloc((size_t)NB * 2 * DD * 4);
  (void)ws_size; (void)in_sizes; (void)n_in; (void)out_size;

  // init state
  zero_f32<<<(NN + 255) / 256, 256, 0, stream>>>((float*)deg, NN);
  zero_f32<<<(NB * DD + 255) / 256, 256, 0, stream>>>(qh, NB * DD);
  zero_f32<<<(NB * DD + 255) / 256, 256, 0, stream>>>(qc, NB * DD);
  zero_f32<<<(NB * 2 * DD + 255) / 256, 256, 0, stream>>>(qstar, NB * 2 * DD);

  // one-time precompute
  edge_hidden<<<NE * HH / 256, 256, 0, stream>>>(ea, nn1_w, nn1_b, Ah);
  prep_b<<<HH * DF / 256, 256, 0, stream>>>(nn2_w, Bm);
  we_gemm<<<dim3(DF / BN, NE / BM), 256, 0, stream>>>(Ah, Bm, nn2_b, We);
  deg_accum<<<NE / 256, 256, 0, stream>>>(ei, deg);
  make_invdeg<<<NN / 256, 256, 0, stream>>>(deg, invd);
  lin0_kernel<<<NN * DD / 256, 256, 0, stream>>>(x, lin0_w, lin0_b, hbuf);
  graph_bounds<<<(NN + 1 + 255) / 256, 256, 0, stream>>>(batch, gstart);

  // 3 message-passing + GRU iterations
  for (int it = 0; it < 3; ++it) {
    zero_f32<<<NN * DD / 256, 256, 0, stream>>>(agg, NN * DD);
    msg_scatter<<<NE / 8, 256, 0, stream>>>(We, hbuf, ei, agg);
    node_update<<<NN, 64, 0, stream>>>(hbuf, agg, invd, root_w, conv_b,
                                       gw_ih, gw_hh, gb_ih, gb_hh);
  }

  // 3 Set2Set steps
  for (int s = 0; s < 3; ++s)
    s2s_step<<<NB, 64, 0, stream>>>(hbuf, gstart, sw_ih, sw_hh, sb_ih, sb_hh,
                                    qh, qc, qstar);

  final_mlp<<<NB, 64, 0, stream>>>(qstar, l1w, l1b, l2w, l2b, y);
}
